// OptimizedCueBasedRNNModel_6262062317723
// MI455X (gfx1250) — matrix operand order repaired and checked by reading, not compiled, not run on hardware
//
#include <hip/hip_runtime.h>
#include <hip/hip_bf16.h>
#include <math.h>

// ---------------- model dims ----------------
#define TT 128
#define BB 64
#define HH 1024
#define EE 1024
#define VV 10001
#define SLP 160            // padded score stride (>= T+1)

typedef __attribute__((ext_vector_type(16))) __bf16 v16bf;
typedef __attribute__((ext_vector_type(8)))  float  v8f;

__device__ __forceinline__ unsigned short f32_to_bf16_rne(float f) {
    unsigned int u = __float_as_uint(f);
    unsigned int r = u + 0x7FFFu + ((u >> 16) & 1u);
    return (unsigned short)(r >> 16);
}
__device__ __forceinline__ float bf16_to_f32(unsigned short h) {
    return __uint_as_float(((unsigned int)h) << 16);
}

// CDNA5 async global->LDS copy (ASYNCcnt-tracked), see CDNA5 ISA
__device__ __forceinline__ void async_ld_b128(unsigned lds_byte_off, const void* gaddr) {
    asm volatile("global_load_async_to_lds_b128 %0, %1, off"
                 :: "v"(lds_byte_off), "v"(gaddr) : "memory");
}

// ---------------- fp32 -> bf16 weight conversion (done once; weights stay L2-resident) ----
__global__ void cvt_f32_bf16_kernel(const float* __restrict__ src,
                                    unsigned short* __restrict__ dst, int n) {
    for (int i = blockIdx.x * blockDim.x + threadIdx.x; i < n; i += gridDim.x * blockDim.x)
        dst[i] = f32_to_bf16_rne(src[i]);
}

// ---------------- init hidden_states[0], Kc[0], Vc[0] ----------------
__global__ void init_state_kernel(const float* __restrict__ h0, const float* __restrict__ k0,
                                  const float* __restrict__ v0, float* __restrict__ out_hid,
                                  float* __restrict__ Kc, float* __restrict__ Vc, int n) {
    for (int i = blockIdx.x * blockDim.x + threadIdx.x; i < n; i += gridDim.x * blockDim.x) {
        out_hid[i] = h0[i];
        Kc[i]      = k0[i];
        Vc[i]      = v0[i];
    }
}

// ---------------- X1(bf16) = [emb_W[obs[t]], hid] : [B, E+H] ----------------
__global__ void build_x1_kernel(const int* __restrict__ obs, const float* __restrict__ embW,
                                const float* __restrict__ hid_prev,
                                unsigned short* __restrict__ X1, int t) {
    const int W = EE + HH;
    for (int i = blockIdx.x * blockDim.x + threadIdx.x; i < BB * W; i += gridDim.x * blockDim.x) {
        int b = i / W, c = i % W;
        float v;
        if (c < EE) {
            int tok = obs[t * BB + b];
            v = embW[(size_t)tok * EE + c];
        } else {
            v = hid_prev[b * HH + (c - EE)];
        }
        X1[i] = f32_to_bf16_rne(v);
    }
}

// ---------------- X2(bf16) = [e, q, attn, hid] : [B, 4H] ----------------
__global__ void build_x2_kernel(const unsigned short* __restrict__ X1,
                                const float* __restrict__ q,
                                const float* __restrict__ attn,
                                unsigned short* __restrict__ X2) {
    const int W = 4 * HH;
    for (int i = blockIdx.x * blockDim.x + threadIdx.x; i < BB * W; i += gridDim.x * blockDim.x) {
        int b = i / W, c = i % W;
        unsigned short v;
        if (c < HH)            v = X1[b * (EE + HH) + c];                              // e
        else if (c < 2 * HH)   v = f32_to_bf16_rne(q[b * HH + (c - HH)]);              // q
        else if (c < 3 * HH)   v = f32_to_bf16_rne(attn[b * HH + (c - 2 * HH)]);       // attn
        else                   v = X1[b * (EE + HH) + EE + (c - 3 * HH)];              // hid
        X2[i] = v;
    }
}

// ---------------- WMMA GEMM: C[M,N] = A_bf16[M,K] * W_bf16[N,K]^T + bias ----------------
// 256 threads = 8 waves; block tile 64x128, K-chunk 64.
// Each wave: 16x64 strip -> 4 f32 accumulators, 8 v_wmma per K-chunk.
// Tiles staged with global_load_async_to_lds_b128, double-buffered, s_wait_asynccnt pipelined.
#define BM 64
#define BN 128
#define BK 64
#define LP 72                      // padded row stride in elems (144B, 16B multiple)
#define AS_ELEMS (2 * BM * LP)
#define WS_ELEMS (2 * BN * LP)

__global__ __launch_bounds__(256) void gemm_bf16_wmma_kernel(
        const unsigned short* __restrict__ A, const unsigned short* __restrict__ W,
        const float* __restrict__ bias, float* __restrict__ C,
        int M, int N, int K) {
    __shared__ unsigned short lds[AS_ELEMS + WS_ELEMS];   // single arena: LDS offset 0

    const int tid  = threadIdx.x;
    const int wave = tid >> 5;
    const int lane = tid & 31;
    const int r16  = lane & 15;
    const int half = lane >> 4;
    const int m0   = (wave >> 1) * 16;
    const int n0   = (wave & 1) * 64;
    const int blockM = blockIdx.y * BM;
    const int blockN = blockIdx.x * BN;

    // loader assignments (uniform control flow so per-wave ASYNCcnt stays 6/tile)
    const int ar = tid >> 2;              // A row 0..63
    const int ak = (tid & 3) * 16;        // A k offset {0,16,32,48}
    const int wr = tid >> 1;              // W row 0..127
    const int wk = (tid & 1) * 32;        // W k offset {0,32}
    const int wn_g   = blockN + wr;
    const bool w_oob = (wn_g >= N);
    const int wclamp = w_oob ? (N - 1) : wn_g;

    auto As_idx = [&](int buf, int r, int k) { return (buf * BM + r) * LP + k; };
    auto Ws_idx = [&](int buf, int r, int k) { return AS_ELEMS + (buf * BN + r) * LP + k; };

    auto issue_tile = [&](int buf, int k0) {
        // A tile: 2 x b128 per thread
        const unsigned short* ap = A + (size_t)(blockM + ar) * K + k0 + ak;
        unsigned oa = (unsigned)(2 * As_idx(buf, ar, ak));
        async_ld_b128(oa,      ap);
        async_ld_b128(oa + 16, ap + 8);
        // W tile: 4 x b128 per thread
        const unsigned short* wp = W + (size_t)wclamp * K + k0 + wk;
        unsigned ow = (unsigned)(2 * Ws_idx(buf, wr, wk));
        async_ld_b128(ow,      wp);
        async_ld_b128(ow + 16, wp + 8);
        async_ld_b128(ow + 32, wp + 16);
        async_ld_b128(ow + 48, wp + 24);
    };

    v8f acc[4] = {{}, {}, {}, {}};
    const int nk = K / BK;

    issue_tile(0, 0);
    for (int i = 0; i < nk; ++i) {
        const int cur = i & 1;
        if (i + 1 < nk) {
            issue_tile(cur ^ 1, (i + 1) * BK);
            asm volatile("s_wait_asynccnt 6" ::: "memory");   // tile i landed, tile i+1 in flight
        } else {
            asm volatile("s_wait_asynccnt 0" ::: "memory");
        }
        if (w_oob) {                                          // zero my (clamped) OOB W rows
            const uint4 z = make_uint4(0u, 0u, 0u, 0u);
            unsigned short* wz = &lds[Ws_idx(cur, wr, wk)];
            *(uint4*)(wz)      = z;
            *(uint4*)(wz + 8)  = z;
            *(uint4*)(wz + 16) = z;
            *(uint4*)(wz + 24) = z;
        }
        __syncthreads();

        #pragma unroll
        for (int ks = 0; ks < BK; ks += 32) {
            // A fragment: lanes 0-15 K={0..15}, lanes 16-31 K={16..31} (+ks)
            v16bf a;
            __builtin_memcpy(&a,             &lds[As_idx(cur, m0 + r16, ks + half * 16)],     16);
            __builtin_memcpy((char*)&a + 16, &lds[As_idx(cur, m0 + r16, ks + half * 16 + 8)], 16);
            #pragma unroll
            for (int j = 0; j < 4; ++j) {
                v16bf b;   // B fragment: lane = column, 16 contiguous K per half-wave
                __builtin_memcpy(&b, &lds[Ws_idx(cur, n0 + 16 * j + r16, ks + half * 16)], 32);
                acc[j] = __builtin_amdgcn_wmma_f32_16x16x32_bf16(false, a, false, b,
                                                                 (short)0, acc[j], false, false);
            }
        }
        __syncthreads();
    }

    // store: VGPR v -> row m0+v (lanes 0-15) / m0+v+8 (lanes 16-31), col = n0 + 16j + lane%16
    #pragma unroll
    for (int j = 0; j < 4; ++j) {
        const int col = blockN + n0 + 16 * j + r16;
        if (col < N) {
            const float bv = bias[col];
            #pragma unroll
            for (int v = 0; v < 8; ++v) {
                const int row = blockM + m0 + v + half * 8;
                C[(size_t)row * N + col] = acc[j][v] + bv;
            }
        }
    }
}

// ---------------- fused LayerNorm + tanh, one block per row ----------------
__global__ __launch_bounds__(256) void ln_tanh_f32_kernel(const float* __restrict__ X,
                                                          const float* __restrict__ g,
                                                          const float* __restrict__ beta,
                                                          float* __restrict__ Y, int N) {
    __shared__ float red[256];
    const int row = blockIdx.x;
    const int tid = threadIdx.x;
    const float* x = X + (size_t)row * N;

    float s = 0.f;
    for (int i = tid; i < N; i += 256) s += x[i];
    red[tid] = s; __syncthreads();
    for (int o = 128; o > 0; o >>= 1) { if (tid < o) red[tid] += red[tid + o]; __syncthreads(); }
    const float mean = red[0] / (float)N;
    __syncthreads();

    float vs = 0.f;
    for (int i = tid; i < N; i += 256) { float d = x[i] - mean; vs += d * d; }
    red[tid] = vs; __syncthreads();
    for (int o = 128; o > 0; o >>= 1) { if (tid < o) red[tid] += red[tid + o]; __syncthreads(); }
    const float rstd = rsqrtf(red[0] / (float)N + 1e-5f);

    for (int i = tid; i < N; i += 256)
        Y[(size_t)row * N + i] = tanhf((x[i] - mean) * rstd * g[i] + beta[i]);
}

__global__ __launch_bounds__(256) void ln_tanh_bf16_kernel(const float* __restrict__ X,
                                                           const float* __restrict__ g,
                                                           const float* __restrict__ beta,
                                                           unsigned short* __restrict__ Y, int N) {
    __shared__ float red[256];
    const int row = blockIdx.x;
    const int tid = threadIdx.x;
    const float* x = X + (size_t)row * N;

    float s = 0.f;
    for (int i = tid; i < N; i += 256) s += x[i];
    red[tid] = s; __syncthreads();
    for (int o = 128; o > 0; o >>= 1) { if (tid < o) red[tid] += red[tid + o]; __syncthreads(); }
    const float mean = red[0] / (float)N;
    __syncthreads();

    float vs = 0.f;
    for (int i = tid; i < N; i += 256) { float d = x[i] - mean; vs += d * d; }
    red[tid] = vs; __syncthreads();
    for (int o = 128; o > 0; o >>= 1) { if (tid < o) red[tid] += red[tid + o]; __syncthreads(); }
    const float rstd = rsqrtf(red[0] / (float)N + 1e-5f);

    for (int i = tid; i < N; i += 256)
        Y[(size_t)row * N + i] = f32_to_bf16_rne(tanhf((x[i] - mean) * rstd * g[i] + beta[i]));
}

// ---------------- attention scores: one dot per wave, wave32 shuffle reduce ----------------
__global__ void scores_kernel(const float* __restrict__ Kc, const float* __restrict__ q,
                              float* __restrict__ scores, int t, float scale) {
    const int pos  = blockIdx.x * 8 + (threadIdx.x >> 5);
    const int b    = blockIdx.y;
    const int lane = threadIdx.x & 31;
    if (pos > t) return;
    const float* kp = Kc + ((size_t)pos * BB + b) * HH;
    const float* qp = q + (size_t)b * HH;
    float s = 0.f;
    for (int i = lane; i < HH; i += 32) s += kp[i] * qp[i];
    #pragma unroll
    for (int o = 16; o > 0; o >>= 1) s += __shfl_xor(s, o, 32);
    if (lane == 0) scores[b * SLP + pos] = s * scale;
}

// ---------------- softmax over pos<=t, then attn = sum_pos w*Vc ----------------
__global__ __launch_bounds__(256) void softmax_attn_kernel(const float* __restrict__ scores,
                                                           const float* __restrict__ Vc,
                                                           float* __restrict__ attn, int t) {
    __shared__ float red[256];
    __shared__ float w[SLP];
    const int b   = blockIdx.x;
    const int tid = threadIdx.x;
    const int P   = t + 1;

    float mx = -3.4e38f;
    for (int p = tid; p < P; p += 256) mx = fmaxf(mx, scores[b * SLP + p]);
    red[tid] = mx; __syncthreads();
    for (int o = 128; o > 0; o >>= 1) { if (tid < o) red[tid] = fmaxf(red[tid], red[tid + o]); __syncthreads(); }
    mx = red[0]; __syncthreads();

    float ls = 0.f;
    for (int p = tid; p < P; p += 256) { float e = expf(scores[b * SLP + p] - mx); w[p] = e; ls += e; }
    red[tid] = ls; __syncthreads();
    for (int o = 128; o > 0; o >>= 1) { if (tid < o) red[tid] += red[tid + o]; __syncthreads(); }
    const float inv = 1.f / red[0];
    __syncthreads();

    for (int h = tid; h < HH; h += 256) {
        float acc = 0.f;
        for (int p = 0; p < P; ++p) acc += w[p] * Vc[((size_t)p * BB + b) * HH + h];
        attn[(size_t)b * HH + h] = acc * inv;
    }
}

// ---------------- fo -> Kc[t+1], Vc[t+1], hidden_states[t+1] (+ bf16 hs for decode) ------
__global__ void scatter_kernel(const float* __restrict__ fo, float* __restrict__ Kc,
                               float* __restrict__ Vc, float* __restrict__ out_hid,
                               unsigned short* __restrict__ hs_bf, int t) {
    const int W = 3 * HH;
    for (int i = blockIdx.x * blockDim.x + threadIdx.x; i < BB * W; i += gridDim.x * blockDim.x) {
        int b = i / W, c = i % W;
        int sec = c / HH, off = c % HH;
        float v = fo[i];
        size_t dst = ((size_t)(t + 1) * BB + b) * HH + off;
        if (sec == 0)      Kc[dst] = v;
        else if (sec == 1) Vc[dst] = v;
        else {
            out_hid[dst] = v;
            hs_bf[((size_t)t * BB + b) * HH + off] = f32_to_bf16_rne(v);
        }
    }
}

// ---------------- host orchestration ----------------
extern "C" void kernel_launch(void* const* d_in, const int* in_sizes, int n_in,
                              void* d_out, int out_size, void* d_ws, size_t ws_size,
                              hipStream_t stream) {
    (void)in_sizes; (void)n_in; (void)out_size; (void)ws_size;

    const int*   obs   = (const int*)  d_in[0];
    const float* h0    = (const float*)d_in[1];
    const float* kci   = (const float*)d_in[2];
    const float* vci   = (const float*)d_in[3];
    const float* embW  = (const float*)d_in[4];
    const float* qW    = (const float*)d_in[5];
    const float* qb    = (const float*)d_in[6];
    const float* qng   = (const float*)d_in[7];
    const float* qnb   = (const float*)d_in[8];
    const float* intW  = (const float*)d_in[9];
    const float* intb  = (const float*)d_in[10];
    const float* intng = (const float*)d_in[11];
    const float* intnb = (const float*)d_in[12];
    const float* finW  = (const float*)d_in[13];
    const float* finb  = (const float*)d_in[14];
    const float* fng   = (const float*)d_in[15];
    const float* fnb   = (const float*)d_in[16];
    const float* decW  = (const float*)d_in[17];
    const float* decb  = (const float*)d_in[18];

    float* out_dec = (float*)d_out;                                  // [T,B,V]
    float* out_hid = out_dec + (size_t)TT * BB * VV;                 // [T+1,B,H]

    // -------- workspace layout --------
    char* ws = (char*)d_ws;
    size_t off = 0;
    auto alloc = [&](size_t bytes) { void* p = ws + off; off = (off + bytes + 255) & ~(size_t)255; return p; };
    unsigned short* WqB  = (unsigned short*)alloc((size_t)HH * (EE + HH) * 2);
    unsigned short* WiB  = (unsigned short*)alloc((size_t)4 * HH * 4 * HH * 2);
    unsigned short* WfB  = (unsigned short*)alloc((size_t)3 * HH * 4 * HH * 2);
    unsigned short* WdB  = (unsigned short*)alloc((size_t)VV * HH * 2);
    unsigned short* hsB  = (unsigned short*)alloc((size_t)TT * BB * HH * 2);
    float* Kc    = (float*)alloc((size_t)(TT + 1) * BB * HH * 4);
    float* Vc    = (float*)alloc((size_t)(TT + 1) * BB * HH * 4);
    unsigned short* X1b  = (unsigned short*)alloc((size_t)BB * (EE + HH) * 2);
    unsigned short* X2b  = (unsigned short*)alloc((size_t)BB * 4 * HH * 2);
    unsigned short* intB = (unsigned short*)alloc((size_t)BB * 4 * HH * 2);
    float* Y1    = (float*)alloc((size_t)BB * HH * 4);
    float* qbuf  = (float*)alloc((size_t)BB * HH * 4);
    float* attn  = (float*)alloc((size_t)BB * HH * 4);
    float* Y2    = (float*)alloc((size_t)BB * 4 * HH * 4);
    float* Y3    = (float*)alloc((size_t)BB * 3 * HH * 4);
    float* fo    = (float*)alloc((size_t)BB * 3 * HH * 4);
    float* sc    = (float*)alloc((size_t)BB * SLP * 4);

    // -------- one-time bf16 weight conversion (L2-resident afterwards) --------
    cvt_f32_bf16_kernel<<<2048, 256, 0, stream>>>(qW,   WqB, HH * (EE + HH));
    cvt_f32_bf16_kernel<<<4096, 256, 0, stream>>>(intW, WiB, 4 * HH * 4 * HH);
    cvt_f32_bf16_kernel<<<4096, 256, 0, stream>>>(finW, WfB, 3 * HH * 4 * HH);
    cvt_f32_bf16_kernel<<<4096, 256, 0, stream>>>(decW, WdB, VV * HH);

    init_state_kernel<<<256, 256, 0, stream>>>(h0, kci, vci, out_hid, Kc, Vc, BB * HH);

    const float scale = 1.0f / sqrtf((float)HH);

    for (int t = 0; t < TT; ++t) {
        const float* hid_prev = out_hid + (size_t)t * BB * HH;

        build_x1_kernel<<<512, 256, 0, stream>>>(obs, embW, hid_prev, X1b, t);

        // q = tanh(LN(X1 @ qW^T + qb))
        gemm_bf16_wmma_kernel<<<dim3(HH / BN, BB / BM), 256, 0, stream>>>(
            X1b, WqB, qb, Y1, BB, HH, EE + HH);
        ln_tanh_f32_kernel<<<BB, 256, 0, stream>>>(Y1, qng, qnb, qbuf, HH);

        // attention over cache positions 0..t
        scores_kernel<<<dim3((t + 1 + 7) / 8, BB), 256, 0, stream>>>(Kc, qbuf, sc, t, scale);
        softmax_attn_kernel<<<BB, 256, 0, stream>>>(sc, Vc, attn, t);

        build_x2_kernel<<<1024, 256, 0, stream>>>(X1b, qbuf, attn, X2b);

        // inter = tanh(LN(X2 @ intW^T + intb))  (bf16 out: next GEMM's A operand)
        gemm_bf16_wmma_kernel<<<dim3(4 * HH / BN, BB / BM), 256, 0, stream>>>(
            X2b, WiB, intb, Y2, BB, 4 * HH, 4 * HH);
        ln_tanh_bf16_kernel<<<BB, 256, 0, stream>>>(Y2, intng, intnb, intB, 4 * HH);

        // fo = tanh(LN(inter @ finW^T + finb))
        gemm_bf16_wmma_kernel<<<dim3(3 * HH / BN, BB / BM), 256, 0, stream>>>(
            intB, WfB, finb, Y3, BB, 3 * HH, 4 * HH);
        ln_tanh_f32_kernel<<<BB, 256, 0, stream>>>(Y3, fng, fnb, fo, 3 * HH);

        scatter_kernel<<<768, 256, 0, stream>>>(fo, Kc, Vc, out_hid, hsB, t);
    }

    // decoded = hs @ decW^T + decb  ([T*B, H] x [H, V])
    gemm_bf16_wmma_kernel<<<dim3((VV + BN - 1) / BN, (TT * BB) / BM), 256, 0, stream>>>(
        hsB, WdB, decb, out_dec, TT * BB, VV, HH);
}
